// CustomGRULayer_65051574665836
// MI455X (gfx1250) — compile-verified, matrix operand order checked by reading, not run on hardware
//
#include <hip/hip_runtime.h>
#include <cmath>

// GRU: T=512, B=64, D=512, H=1024
#define T_STEPS 512
#define B_DIM   64
#define D_DIM   512
#define H_DIM   1024

typedef __attribute__((ext_vector_type(16))) __bf16 v16bf;
typedef __attribute__((ext_vector_type(8)))  __bf16 v8bf;
typedef __attribute__((ext_vector_type(8)))  float  v8f;

// ---------------------------------------------------------------------------
// WMMA 16x16x32 bf16 fragment index math (wave32, CDNA5 ISA 7.12.2):
//  A (16x32, MxK): lanes 0-15 hold M=lane, K in {0..7}∪{16..23};
//                  lanes 16-31 hold M=lane-16, K in {8..15}∪{24..31}.
//  Per-lane v16bf element j: K = (j/8)*16 + (lane>=16 ? 8 : 0) + (j%8)
//  B (32x16, KxN) mirrored with N%16 in place of M.
//  One fragment chunk = 32 lanes * 16 bf16 = 512 elements, stored contiguous.
//  NOTE: 8 consecutive 8-aligned K values share (lane,hi,grp) and occupy 8
//  contiguous j slots -> 16 contiguous bytes (enables vectorized staging).
// ---------------------------------------------------------------------------
__device__ __forceinline__ int frag_pos(int rowlane /* m or n%16 */, int kk /* k%32 */) {
    int rem = kk & 15;
    int hi  = rem >> 3;
    int wi  = kk & 7;
    int grp = kk >> 4;
    int lane = rowlane + (hi << 4);
    int j    = (grp << 3) + wi;
    return lane * 16 + j;
}

// Convert three [K,H] fp32 weight matrices into fragment-ordered bf16:
// dst[(((g*(H/16) + ntile)*(K/32) + kc))*512 + frag_pos(n%16, k%32)]
__global__ __launch_bounds__(256) void cvt_frag_kernel(
    const float* __restrict__ S0, const float* __restrict__ S1,
    const float* __restrict__ S2, __bf16* __restrict__ dst, int K)
{
    int idx = blockIdx.x * 256 + threadIdx.x;
    int total = 3 * K * H_DIM;
    if (idx >= total) return;
    int g  = idx / (K * H_DIM);
    int r2 = idx - g * K * H_DIM;
    int k  = r2 / H_DIM;
    int n  = r2 - k * H_DIM;
    const float* S = (g == 0) ? S0 : ((g == 1) ? S1 : S2);
    float v = S[(size_t)k * H_DIM + n];
    int ntile = n >> 4, nl = n & 15;
    int kc = k >> 5,  kk = k & 31;
    int KC = K >> 5;
    size_t off = ((size_t)((g * (H_DIM / 16) + ntile) * KC + kc)) * 512
               + frag_pos(nl, kk);
    dst[off] = (__bf16)v;
}

__global__ __launch_bounds__(256) void init_h_kernel(
    const float* __restrict__ hidden, float* __restrict__ h0)
{
    int i = blockIdx.x * 256 + threadIdx.x;
    if (i < B_DIM * H_DIM) h0[i] = hidden[i];
}

__global__ __launch_bounds__(256) void copy_kernel(
    const float* __restrict__ src, float* __restrict__ dst, int n)
{
    int i = blockIdx.x * 256 + threadIdx.x;
    if (i < n) dst[i] = src[i];
}

#define WMMA_BF16(A, Bv, C) \
    __builtin_amdgcn_wmma_f32_16x16x32_bf16(false, (A), false, (Bv), (short)0, (C), false, false)

// Vectorized fp32 -> bf16 A-fragment staging: one 8-aligned K-octet = one
// 32B coalesced global read + one 16B ds_store_b128.
__device__ __forceinline__ void stage_octet(
    const float* __restrict__ srcrow, __bf16* __restrict__ ldsbuf,
    int m, int k8)
{
    const float4* p = (const float4*)(srcrow + k8);
    float4 f0 = p[0];
    float4 f1 = p[1];
    v8bf v;
    v[0] = (__bf16)f0.x; v[1] = (__bf16)f0.y;
    v[2] = (__bf16)f0.z; v[3] = (__bf16)f0.w;
    v[4] = (__bf16)f1.x; v[5] = (__bf16)f1.y;
    v[6] = (__bf16)f1.z; v[7] = (__bf16)f1.w;
    int kc  = k8 >> 5;
    int kk  = k8 & 31;                 // 0, 8, 16, 24
    int hi  = (kk & 15) >> 3;
    int grp = kk >> 4;
    int lane = m + (hi << 4);
    int jb   = grp << 3;
    *(v8bf*)&ldsbuf[(kc * 32 + lane) * 16 + jb] = v;
}

// One recurrence step: gates = x[t]@W + h@U (+bias), GRU update, write h_new.
// Grid: 32 blocks x 256 threads (8 waves). Block = 1 M-tile x 8 N-tiles.
__global__ __launch_bounds__(256) void gru_step_kernel(
    const float*  __restrict__ x_t,    // [B, D]  fp32 (inputs + t*B*D)
    const float*  __restrict__ h_in,   // [B, H]  fp32
    float*        __restrict__ h_out,  // [B, H]  fp32
    const __bf16* __restrict__ Wfrag,  // gates r,z,h ; K=D
    const __bf16* __restrict__ Ufrag,  // gates r,z,h ; K=H
    const float*  __restrict__ br, const float* __restrict__ bz,
    const float*  __restrict__ bh,
    float*        __restrict__ out_t)  // [B, H]  fp32 (outputs + t*B*H)
{
    __shared__ __bf16 ldsx[16 * D_DIM];   // 16 KB: A-fragments of x[t] rows
    __shared__ __bf16 ldsh[16 * H_DIM];   // 32 KB: A-fragments of h rows

    const int tid    = threadIdx.x;
    const int mtile  = blockIdx.x & 3;          // 4 M-tiles (B=64)
    const int ntbase = (blockIdx.x >> 2) * 8;   // 8 N-tiles per block

    // Stage x[t] (16 x 512): 16*64 = 1024 K-octets, 4 per thread
#pragma unroll
    for (int c = tid; c < 16 * (D_DIM / 8); c += 256) {
        int m  = c >> 6;                 // c / (D_DIM/8)
        int k8 = (c & 63) << 3;
        stage_octet(x_t + (size_t)(mtile * 16 + m) * D_DIM, ldsx, m, k8);
    }
    // Stage h (16 x 1024): 16*128 = 2048 K-octets, 8 per thread
#pragma unroll
    for (int c = tid; c < 16 * (H_DIM / 8); c += 256) {
        int m  = c >> 7;                 // c / (H_DIM/8)
        int k8 = (c & 127) << 3;
        stage_octet(h_in + (size_t)(mtile * 16 + m) * H_DIM, ldsh, m, k8);
    }
    __syncthreads();

    const int wave  = tid >> 5;
    const int lane  = tid & 31;
    const int ntile = ntbase + wave;

    const v16bf* LX = (const v16bf*)ldsx;
    const v16bf* LH = (const v16bf*)ldsh;
    const v16bf* WF = (const v16bf*)Wfrag;
    const v16bf* UF = (const v16bf*)Ufrag;

    v8f ar = {}, az = {}, au = {}, axh = {};

    // x-projection part: K = D = 512 -> 16 k-chunks, 3 gate chains interleaved
#pragma unroll 2
    for (int kc = 0; kc < D_DIM / 32; ++kc) {
        v16bf a  = LX[kc * 32 + lane];
        v16bf b0 = WF[(size_t)((0 * 64 + ntile) * 16 + kc) * 32 + lane];
        v16bf b1 = WF[(size_t)((1 * 64 + ntile) * 16 + kc) * 32 + lane];
        v16bf b2 = WF[(size_t)((2 * 64 + ntile) * 16 + kc) * 32 + lane];
        ar  = WMMA_BF16(a, b0, ar);
        az  = WMMA_BF16(a, b1, az);
        axh = WMMA_BF16(a, b2, axh);
    }
    // recurrent part: K = H = 1024 -> 32 k-chunks
#pragma unroll 2
    for (int kc = 0; kc < H_DIM / 32; ++kc) {
        v16bf a  = LH[kc * 32 + lane];
        v16bf b0 = UF[(size_t)((0 * 64 + ntile) * 32 + kc) * 32 + lane];
        v16bf b1 = UF[(size_t)((1 * 64 + ntile) * 32 + kc) * 32 + lane];
        v16bf b2 = UF[(size_t)((2 * 64 + ntile) * 32 + kc) * 32 + lane];
        ar = WMMA_BF16(a, b0, ar);
        az = WMMA_BF16(a, b1, az);
        au = WMMA_BF16(a, b2, au);
    }

    // C/D layout: VGPR r, lanes 0-15 -> M=r, lanes 16-31 -> M=8+r; N = lane%16
    const int l16 = lane & 15, lhi = lane >> 4;
    const int n   = ntile * 16 + l16;
    const float brv = br[n], bzv = bz[n], bhv = bh[n];

#pragma unroll
    for (int r = 0; r < 8; ++r) {
        int b = mtile * 16 + lhi * 8 + r;
        float rg = 1.0f / (1.0f + __expf(-(ar[r] + brv)));
        float zg = 1.0f / (1.0f + __expf(-(az[r] + bzv)));
        float hh = tanhf(axh[r] + bhv + rg * au[r]);
        float ho = h_in[(size_t)b * H_DIM + n];
        float hn = (1.0f - zg) * ho + zg * hh;
        h_out[(size_t)b * H_DIM + n] = hn;
        out_t[(size_t)b * H_DIM + n] = hn;
    }
}

extern "C" void kernel_launch(void* const* d_in, const int* in_sizes, int n_in,
                              void* d_out, int out_size, void* d_ws, size_t ws_size,
                              hipStream_t stream) {
    const float* inputs = (const float*)d_in[0];
    const float* hidden = (const float*)d_in[1];
    const float* Wz     = (const float*)d_in[2];
    const float* Uz     = (const float*)d_in[3];
    const float* bz     = (const float*)d_in[4];
    const float* Wr     = (const float*)d_in[5];
    const float* Ur     = (const float*)d_in[6];
    const float* br     = (const float*)d_in[7];
    const float* Wh     = (const float*)d_in[8];
    const float* Uh     = (const float*)d_in[9];
    const float* bh     = (const float*)d_in[10];
    float* out = (float*)d_out;

    // Workspace layout (~9.5 MB total):
    //   Wfrag : 3*512*1024  bf16 = 3 MB
    //   Ufrag : 3*1024*1024 bf16 = 6 MB
    //   h ping/pong : 2 * 64*1024 fp32 = 512 KB
    char* ws = (char*)d_ws;
    __bf16* Wfrag = (__bf16*)ws;
    __bf16* Ufrag = (__bf16*)(ws + (size_t)3 * D_DIM * H_DIM * 2);
    float*  h0    = (float*)(ws + (size_t)3 * D_DIM * H_DIM * 2
                                + (size_t)3 * H_DIM * H_DIM * 2);
    float*  h1    = h0 + B_DIM * H_DIM;

    // One-time (per launch, deterministic) weight fragment conversion.
    {
        int totW = 3 * D_DIM * H_DIM;
        cvt_frag_kernel<<<(totW + 255) / 256, 256, 0, stream>>>(Wr, Wz, Wh, Wfrag, D_DIM);
        int totU = 3 * H_DIM * H_DIM;
        cvt_frag_kernel<<<(totU + 255) / 256, 256, 0, stream>>>(Ur, Uz, Uh, Ufrag, H_DIM);
    }
    init_h_kernel<<<(B_DIM * H_DIM + 255) / 256, 256, 0, stream>>>(hidden, h0);

    float* bufs[2] = { h0, h1 };
    for (int t = 0; t < T_STEPS; ++t) {
        const float* hin  = bufs[t & 1];
        float*       hout = bufs[(t + 1) & 1];
        gru_step_kernel<<<32, 256, 0, stream>>>(
            inputs + (size_t)t * B_DIM * D_DIM,
            hin, hout, Wfrag, Ufrag, br, bz, bh,
            out + (size_t)t * B_DIM * H_DIM);
    }
    // h_last ([1,B,H]) appended after outputs ([T,B,H])
    copy_kernel<<<(B_DIM * H_DIM + 255) / 256, 256, 0, stream>>>(
        bufs[T_STEPS & 1], out + (size_t)T_STEPS * B_DIM * H_DIM, B_DIM * H_DIM);
}
